// DistilBertEncoder_63007170232935
// MI455X (gfx1250) — compile-verified, matrix operand order checked by reading, not run on hardware
//
#include <hip/hip_runtime.h>
#include <hip/hip_bf16.h>

#define LAYERS 6
#define BB 8
#define SS 512
#define DD 768
#define FF 3072
#define NH 12
#define HD 64
#define MM (BB * SS) /* 4096 tokens */

typedef unsigned int u32;
typedef __attribute__((ext_vector_type(8)))  float  v8f;
typedef __attribute__((ext_vector_type(16))) __bf16 v16bf;
typedef __attribute__((ext_vector_type(8)))  __bf16 v8bf;

// ---------------------------------------------------------------- helpers ---

__device__ __forceinline__ __bf16 f2bf(float f) {
  u32 u = __float_as_uint(f);
  u32 r = (u + 0x7FFFu + ((u >> 16) & 1u)) >> 16;  // round-to-nearest-even
  unsigned short s = (unsigned short)r;
  return __builtin_bit_cast(__bf16, s);
}

__device__ __forceinline__ v8f wmma_bf(v16bf a, v16bf b, v8f c) {
  return __builtin_amdgcn_wmma_f32_16x16x32_bf16(false, a, false, b, (short)0,
                                                 c, false, false);
}

// Load one 16x32 bf16 A/B fragment row for this lane.  ISA 7.12.2: lane<16
// holds K {0..7,16..23} of row (lane&15); lane>=16 holds K {8..15,24..31}.
// `base` points at element [row, k0] of a row-major [*,K] matrix (row/k0
// already applied by the caller).
__device__ __forceinline__ v16bf load_frag(const __bf16* base, int lane) {
  const __bf16* p = base + ((lane >> 4) << 3);
  v8bf lo = *reinterpret_cast<const v8bf*>(p);
  v8bf hi = *reinterpret_cast<const v8bf*>(p + 16);
  v16bf r;
#pragma unroll
  for (int i = 0; i < 8; ++i) { r[i] = lo[i]; r[i + 8] = hi[i]; }
  return r;
}

// ------------------------------------------------ radix select (k-th stat) ---
// Finds the element of ascending rank n/2 (the TopK threshold for
// THRESHOLD=0.5) of each of 3 score matrices via 4 MSB-first histogram passes
// over order-preserving uint keys.

__global__ void k_sel_init(u32* state, u32* hist, u32 rank) {
  int m = blockIdx.x;
  hist[m * 256 + threadIdx.x] = 0;
  if (threadIdx.x == 0) { state[m * 2] = 0; state[m * 2 + 1] = rank; }
}

__global__ void k_sel_hist(const float* __restrict__ scores, u32 n, int shift,
                           const u32* __restrict__ state, u32* __restrict__ hist) {
  int m = blockIdx.y;
  const float* s = scores + (size_t)m * n;
  u32 prefix = state[m * 2];
  u32 mask = (shift == 24) ? 0u : (0xFFFFFFFFu << (shift + 8));
  u32* h = hist + m * 256;
  for (u32 i = blockIdx.x * blockDim.x + threadIdx.x; i < n;
       i += gridDim.x * blockDim.x) {
    u32 u = __float_as_uint(s[i]);
    u32 key = (u & 0x80000000u) ? ~u : (u | 0x80000000u);
    if ((key & mask) == prefix) atomicAdd(&h[(key >> shift) & 0xFFu], 1u);
  }
}

__global__ void k_sel_scan(int shift, u32* state, u32* hist,
                           float* thr, int thrOfs) {
  int m = blockIdx.x;
  u32* h = hist + m * 256;
  u32 r = state[m * 2 + 1];
  u32 cum = 0;
  int bin = 255;
  for (int i = 0; i < 256; ++i) {
    u32 c = h[i];
    if (cum + c > r) { bin = i; break; }
    cum += c;
  }
  state[m * 2]     |= ((u32)bin) << shift;
  state[m * 2 + 1]  = r - cum;
  for (int i = 0; i < 256; ++i) h[i] = 0;
  if (shift == 0) {
    u32 key = state[m * 2];
    u32 u = (key & 0x80000000u) ? (key ^ 0x80000000u) : ~key;
    thr[thrOfs + m] = __uint_as_float(u);
  }
}

// ------------------------------------------------------- weight preparation ---

__global__ void k_maskw(const float* __restrict__ W, const float* __restrict__ ms,
                        const float* __restrict__ thr, int thrOfs,
                        __bf16* __restrict__ mw, size_t OK) {
  size_t total = 3 * OK;
  for (size_t i = (size_t)blockIdx.x * blockDim.x + threadIdx.x; i < total;
       i += (size_t)gridDim.x * blockDim.x) {
    size_t m = i / OK, rem = i - m * OK;
    mw[i] = (ms[i] >= thr[thrOfs + m]) ? f2bf(W[rem]) : f2bf(0.0f);
  }
}

__global__ void k_cvt_bf(const float* __restrict__ x, __bf16* __restrict__ y,
                         size_t n) {
  for (size_t i = (size_t)blockIdx.x * blockDim.x + threadIdx.x; i < n;
       i += (size_t)gridDim.x * blockDim.x)
    y[i] = f2bf(x[i]);
}

__global__ void k_gelu_bf(const float* __restrict__ x, __bf16* __restrict__ y,
                          size_t n) {
  for (size_t i = (size_t)blockIdx.x * blockDim.x + threadIdx.x; i < n;
       i += (size_t)gridDim.x * blockDim.x) {
    float v = x[i];
    v = 0.5f * v * (1.0f + erff(v * 0.70710678118654752f));
    y[i] = f2bf(v);
  }
}

// ----------------------------------------------------------------- gating ---

__global__ __launch_bounds__(256) void k_gate3(const float* __restrict__ x,
                                               const float* __restrict__ gw,
                                               const float* __restrict__ gb,
                                               unsigned char* __restrict__ eo) {
  __shared__ float red[3][256];
  int t = blockIdx.x, tid = threadIdx.x;
  float s0 = 0.f, s1 = 0.f, s2 = 0.f;
  for (int c = tid; c < DD; c += 256) {
    float xv = x[(size_t)t * DD + c];
    s0 += xv * gw[c];
    s1 += xv * gw[DD + c];
    s2 += xv * gw[2 * DD + c];
  }
  red[0][tid] = s0; red[1][tid] = s1; red[2][tid] = s2;
  __syncthreads();
  for (int o = 128; o; o >>= 1) {
    if (tid < o) {
      red[0][tid] += red[0][tid + o];
      red[1][tid] += red[1][tid + o];
      red[2][tid] += red[2][tid + o];
    }
    __syncthreads();
  }
  if (tid == 0) {
    float g0 = red[0][0] + gb[0], g1 = red[1][0] + gb[1], g2 = red[2][0] + gb[2];
    int e = 0; float m = g0;
    if (g1 > m) { m = g1; e = 1; }
    if (g2 > m) { e = 2; }
    eo[t] = (unsigned char)e;
  }
}

// ------------------------------------------------------------ MoE bf16 GEMM ---
// C[4096,N] = A[4096,K] * W[e,N,K]^T (+bias), per-row expert select.
// Block = 8 waves sharing one 16-col N tile over a 512-row M stripe; each wave
// owns 64 M rows (4 accumulator tiles x NEXP experts).  The shared B tile
// (NEXP x 16 x 32 bf16) is streamed into LDS once per k-step with gfx1250
// GLOBAL_LOAD_ASYNC_TO_LDS_B128 (ASYNCcnt), then read back as WMMA fragments
// with ds_load_b128.

template <int NEXP>
__global__ __launch_bounds__(256) void k_gemm(
    const __bf16* __restrict__ A, const __bf16* __restrict__ W,
    const float* __restrict__ bias, const unsigned char* __restrict__ eid,
    float* __restrict__ Cf, __bf16* __restrict__ Cb, int N, int K) {
  __shared__ __bf16 sB[3 * 16 * 32];  // up to 3 experts x 16 rows x 32 k
  int lane = threadIdx.x & 31, wave = threadIdx.x >> 5;
  int lr = lane & 15;
  int sel = (lane >> 4) << 3;
  int tilesN = N >> 4;
  int tn = blockIdx.x % tilesN;
  int mblk = blockIdx.x / tilesN;
  int mbase = mblk * 512 + wave * 64;
  size_t estride = (size_t)N * K;

  const __bf16* ap[4];
#pragma unroll
  for (int s = 0; s < 4; ++s)
    ap[s] = A + (size_t)(mbase + s * 16 + lr) * K;
  const __bf16* bp = W + (size_t)(tn * 16) * K;

  // Per-lane async-copy coordinates: 64 chunks of 16B per expert tile;
  // lane L of instruction j covers chunk j*32+L -> row (c>>2), 16B sub (c&3).
  int brow0 = lane >> 2, bchunk = lane & 3;
  int brow1 = 8 + (lane >> 2);
  unsigned ldsBase = (unsigned)(uintptr_t)(&sB[0]);  // wave-relative LDS offset

  v8f acc[4][NEXP] = {};
  for (int k0 = 0; k0 < K; k0 += 32) {
    if (wave == 0) {
#pragma unroll
      for (int e = 0; e < NEXP; ++e) {
        const __bf16* src = bp + (size_t)e * estride + k0;
        unsigned long long g0 =
            (unsigned long long)(uintptr_t)(src + (size_t)brow0 * K + bchunk * 8);
        unsigned long long g1 =
            (unsigned long long)(uintptr_t)(src + (size_t)brow1 * K + bchunk * 8);
        unsigned l0 = ldsBase + (unsigned)(e * 1024 + brow0 * 64 + bchunk * 16);
        unsigned l1 = ldsBase + (unsigned)(e * 1024 + brow1 * 64 + bchunk * 16);
        asm volatile("global_load_async_to_lds_b128 %0, %1, off"
                     :: "v"(l0), "v"(g0) : "memory");
        asm volatile("global_load_async_to_lds_b128 %0, %1, off"
                     :: "v"(l1), "v"(g1) : "memory");
      }
      asm volatile("s_wait_asynccnt 0" ::: "memory");
    }
    __syncthreads();
    v16bf bfr[NEXP];
#pragma unroll
    for (int e = 0; e < NEXP; ++e)
      bfr[e] = load_frag(&sB[e * 512 + lr * 32], lane);
#pragma unroll
    for (int s = 0; s < 4; ++s) {
      __builtin_prefetch(ap[s] + k0 + 512, 0, 1);  // global_prefetch_b8
      v16bf af = load_frag(ap[s] + k0, lane);
#pragma unroll
      for (int e = 0; e < NEXP; ++e)
        acc[s][e] = wmma_bf(af, bfr[e], acc[s][e]);
    }
    __syncthreads();
  }

  int col = tn * 16 + lr;
  float bc = bias[col];
#pragma unroll
  for (int s = 0; s < 4; ++s) {
#pragma unroll
    for (int r = 0; r < 8; ++r) {
      int row = mbase + s * 16 + r + sel;
      float v;
      if (NEXP == 1) {
        v = acc[s][0][r];
      } else {
        int e = eid[row];
        v = (e == 0) ? acc[s][0][r] : ((e == 1) ? acc[s][1][r] : acc[s][2][r]);
      }
      v += bc;
      size_t o = (size_t)row * N + col;
      if (Cb) Cb[o] = f2bf(v); else Cf[o] = v;
    }
  }
}

// -------------------------------------------------------------- V transpose ---
// v_bf[(b,s),(h,d)] -> vt[(b,h),d,s]  so P*V B-fragments are K(=s)-contiguous.

__global__ void k_vt(const __bf16* __restrict__ vb, __bf16* __restrict__ vt) {
  size_t total = (size_t)BB * NH * HD * SS;
  for (size_t i = (size_t)blockIdx.x * blockDim.x + threadIdx.x; i < total;
       i += (size_t)gridDim.x * blockDim.x) {
    int s  = (int)(i & (SS - 1));
    int d  = (int)((i >> 9) & (HD - 1));
    int bh = (int)(i >> 15);
    int b = bh / NH, h = bh - b * NH;
    vt[i] = vb[((size_t)(b * SS + s)) * DD + h * HD + d];
  }
}

// ---------------------------------------------------------------- attention ---
// One wave per (b, h, 16-row q tile).  QK^T + softmax (f32 in LDS) + P*V,
// all matmuls through v_wmma_f32_16x16x32_bf16.

__global__ __launch_bounds__(32) void k_attn(
    const __bf16* __restrict__ qb, const __bf16* __restrict__ kb,
    const __bf16* __restrict__ vt, const float* __restrict__ amask,
    const float* __restrict__ hmask, __bf16* __restrict__ ctxb) {
  __shared__ float sS[16 * SS];
  __shared__ float sRow[16];
  int qt = blockIdx.x, bh = blockIdx.y;
  int b = bh / NH, h = bh - b * NH;
  int lane = threadIdx.x;
  int lr = lane & 15, sel = (lane >> 4) << 3;
  const float scale = 0.125f;  // 1/sqrt(64)

  const __bf16* qbase = qb + ((size_t)(b * SS + qt * 16 + lr)) * DD + h * HD;
  v16bf qf0 = load_frag(qbase, lane);
  v16bf qf1 = load_frag(qbase + 32, lane);

  for (int kt = 0; kt < SS / 16; ++kt) {
    const __bf16* kbase = kb + ((size_t)(b * SS + kt * 16 + lr)) * DD + h * HD;
    v16bf kf0 = load_frag(kbase, lane);
    v16bf kf1 = load_frag(kbase + 32, lane);
    v8f acc = {};
    acc = wmma_bf(qf0, kf0, acc);
    acc = wmma_bf(qf1, kf1, acc);
    int col = kt * 16 + lr;
    float am = amask[b * SS + col];
#pragma unroll
    for (int r = 0; r < 8; ++r)
      sS[(r + sel) * SS + col] = acc[r] * scale + am;
  }
  __syncthreads();

  float hm = hmask[h];
  for (int r = 0; r < 16; ++r) {
    float mx = -3.0e38f;
    for (int c = lane; c < SS; c += 32) mx = fmaxf(mx, sS[r * SS + c]);
#pragma unroll
    for (int off = 16; off; off >>= 1) mx = fmaxf(mx, __shfl_xor(mx, off, 32));
    float sum = 0.0f;
    for (int c = lane; c < SS; c += 32) {
      float e = __expf(sS[r * SS + c] - mx);
      sS[r * SS + c] = e;
      sum += e;
    }
#pragma unroll
    for (int off = 16; off; off >>= 1) sum += __shfl_xor(sum, off, 32);
    if (lane == 0) sRow[r] = hm / sum;
  }
  __syncthreads();

  v8f acc[4] = {};
  const __bf16* vbase0 = vt + (size_t)bh * HD * SS;
  float rs = sRow[lr];
  for (int k0 = 0; k0 < SS; k0 += 32) {
    v16bf pf;
    int kb0 = k0 + sel;
#pragma unroll
    for (int i = 0; i < 8; ++i) {
      pf[i]     = f2bf(sS[lr * SS + kb0 + i] * rs);
      pf[i + 8] = f2bf(sS[lr * SS + kb0 + 16 + i] * rs);
    }
#pragma unroll
    for (int dt = 0; dt < 4; ++dt) {
      v16bf vf = load_frag(vbase0 + (size_t)(dt * 16 + lr) * SS + k0, lane);
      acc[dt] = wmma_bf(pf, vf, acc[dt]);
    }
  }
#pragma unroll
  for (int dt = 0; dt < 4; ++dt) {
#pragma unroll
    for (int r = 0; r < 8; ++r) {
      int row = qt * 16 + r + sel;
      ctxb[((size_t)(b * SS + row)) * DD + h * HD + dt * 16 + lr] =
          f2bf(acc[dt][r]);
    }
  }
}

// ---------------------------------------------------------------- layernorm ---

__global__ __launch_bounds__(256) void k_ln(
    const float* __restrict__ a, const float* __restrict__ res,
    const float* __restrict__ g, const float* __restrict__ be,
    float* __restrict__ y, __bf16* __restrict__ ybf) {
  __shared__ float red[256];
  int row = blockIdx.x, t = threadIdx.x;
  size_t base = (size_t)row * DD;
  float x0 = a[base + t]       + res[base + t];
  float x1 = a[base + t + 256] + res[base + t + 256];
  float x2 = a[base + t + 512] + res[base + t + 512];
  red[t] = x0 + x1 + x2;
  __syncthreads();
  for (int o = 128; o; o >>= 1) { if (t < o) red[t] += red[t + o]; __syncthreads(); }
  float mu = red[0] * (1.0f / DD);
  __syncthreads();
  float d0 = x0 - mu, d1 = x1 - mu, d2 = x2 - mu;
  red[t] = d0 * d0 + d1 * d1 + d2 * d2;
  __syncthreads();
  for (int o = 128; o; o >>= 1) { if (t < o) red[t] += red[t + o]; __syncthreads(); }
  float rstd = rsqrtf(red[0] * (1.0f / DD) + 1e-12f);
  float v0 = d0 * rstd * g[t]       + be[t];
  float v1 = d1 * rstd * g[t + 256] + be[t + 256];
  float v2 = d2 * rstd * g[t + 512] + be[t + 512];
  y[base + t] = v0; y[base + t + 256] = v1; y[base + t + 512] = v2;
  if (ybf) {
    ybf[base + t] = f2bf(v0);
    ybf[base + t + 256] = f2bf(v1);
    ybf[base + t + 512] = f2bf(v2);
  }
}

// ------------------------------------------------------------------- launch ---

extern "C" void kernel_launch(void* const* d_in, const int* in_sizes, int n_in,
                              void* d_out, int out_size, void* d_ws,
                              size_t ws_size, hipStream_t stream) {
  (void)in_sizes; (void)n_in; (void)out_size; (void)ws_size;
  const float* hidden = (const float*)d_in[0];
  const float* amask  = (const float*)d_in[1];
  const float* hmask  = (const float*)d_in[2];
  const float* qW  = (const float*)d_in[3];
  const float* qb_ = (const float*)d_in[4];
  const float* qms = (const float*)d_in[5];
  const float* qgw = (const float*)d_in[6];
  const float* qgb = (const float*)d_in[7];
  const float* kW  = (const float*)d_in[8];
  const float* kb_ = (const float*)d_in[9];
  const float* kms = (const float*)d_in[10];
  const float* kgw = (const float*)d_in[11];
  const float* kgb = (const float*)d_in[12];
  const float* vW  = (const float*)d_in[13];
  const float* vb_ = (const float*)d_in[14];
  const float* vms = (const float*)d_in[15];
  const float* vgw = (const float*)d_in[16];
  const float* vgb = (const float*)d_in[17];
  const float* aoW = (const float*)d_in[18];
  const float* aob = (const float*)d_in[19];
  const float* ln1g = (const float*)d_in[20];
  const float* ln1b = (const float*)d_in[21];
  const float* iW  = (const float*)d_in[22];
  const float* ib_ = (const float*)d_in[23];
  const float* ims = (const float*)d_in[24];
  const float* igw = (const float*)d_in[25];
  const float* igb = (const float*)d_in[26];
  const float* oW  = (const float*)d_in[27];
  const float* ob_ = (const float*)d_in[28];
  const float* ln2g = (const float*)d_in[29];
  const float* ln2b = (const float*)d_in[30];

  // Workspace carve-up.
  char* wsb = (char*)d_ws;
  size_t cur = 0;
  auto take = [&](size_t bytes) -> void* {
    void* p = wsb + cur;
    cur = (cur + bytes + 255) & ~(size_t)255;
    return p;
  };
  u32*   selState = (u32*)take(6 * 4);
  u32*   hist     = (u32*)take(3 * 256 * 4);
  float* thr      = (float*)take(12 * 4);
  unsigned char* expq = (unsigned char*)take(MM);
  unsigned char* expk = (unsigned char*)take(MM);
  unsigned char* expv = (unsigned char*)take(MM);
  unsigned char* expi = (unsigned char*)take(MM);
  __bf16* h_bf   = (__bf16*)take((size_t)MM * DD * 2);
  __bf16* ao_bf  = (__bf16*)take((size_t)MM * DD * 2);
  __bf16* q_bf   = (__bf16*)take((size_t)MM * DD * 2);
  __bf16* kk_bf  = (__bf16*)take((size_t)MM * DD * 2);
  __bf16* v_bf   = (__bf16*)take((size_t)MM * DD * 2);
  __bf16* vtb    = (__bf16*)take((size_t)MM * DD * 2);
  __bf16* ctx_bf = (__bf16*)take((size_t)MM * DD * 2);
  __bf16* int_bf = (__bf16*)take((size_t)MM * FF * 2);
  __bf16* mwq    = (__bf16*)take((size_t)3 * DD * DD * 2);
  __bf16* mwk    = (__bf16*)take((size_t)3 * DD * DD * 2);
  __bf16* mwv    = (__bf16*)take((size_t)3 * DD * DD * 2);
  __bf16* mwi    = (__bf16*)take((size_t)3 * FF * DD * 2);
  __bf16* wao    = (__bf16*)take((size_t)DD * DD * 2);
  __bf16* wo     = (__bf16*)take((size_t)DD * FF * 2);
  float*  tmp    = (float*)take((size_t)MM * DD * 4);
  float*  attn   = (float*)take((size_t)MM * DD * 4);
  float*  inter  = (float*)take((size_t)MM * FF * 4);
  float*  hcur   = (float*)take((size_t)MM * DD * 4);

  const int gemmD = (MM / 512) * (DD / 16);  // 384 blocks
  const int gemmF = (MM / 512) * (FF / 16);  // 1536 blocks

  for (int l = 0; l < LAYERS; ++l) {
    const float* hin = (l == 0) ? hidden : hcur;
    k_cvt_bf<<<2048, 256, 0, stream>>>(hin, h_bf, (size_t)MM * DD);

    struct PSpec {
      const float *W, *b, *ms, *gw, *gb;
      __bf16* mw; unsigned char* ei; __bf16* out; int to;
    } P[3] = {
      {qW + (size_t)l * DD * DD, qb_ + (size_t)l * DD, qms + (size_t)l * 3 * DD * DD,
       qgw + (size_t)l * 3 * DD, qgb + (size_t)l * 3, mwq, expq, q_bf, 0},
      {kW + (size_t)l * DD * DD, kb_ + (size_t)l * DD, kms + (size_t)l * 3 * DD * DD,
       kgw + (size_t)l * 3 * DD, kgb + (size_t)l * 3, mwk, expk, kk_bf, 3},
      {vW + (size_t)l * DD * DD, vb_ + (size_t)l * DD, vms + (size_t)l * 3 * DD * DD,
       vgw + (size_t)l * 3 * DD, vgb + (size_t)l * 3, mwv, expv, v_bf, 6},
    };

    for (int p = 0; p < 3; ++p) {
      k_sel_init<<<3, 256, 0, stream>>>(selState, hist, (u32)(DD * DD / 2));
      for (int shift = 24; shift >= 0; shift -= 8) {
        k_sel_hist<<<dim3(512, 3), 256, 0, stream>>>(P[p].ms, (u32)(DD * DD),
                                                     shift, selState, hist);
        k_sel_scan<<<3, 1, 0, stream>>>(shift, selState, hist, thr, P[p].to);
      }
      k_maskw<<<2048, 256, 0, stream>>>(P[p].W, P[p].ms, thr, P[p].to, P[p].mw,
                                        (size_t)DD * DD);
      k_gate3<<<MM, 256, 0, stream>>>(hin, P[p].gw, P[p].gb, P[p].ei);
      k_gemm<3><<<gemmD, 256, 0, stream>>>(h_bf, P[p].mw, P[p].b, P[p].ei,
                                           nullptr, P[p].out, DD, DD);
    }

    // intermediate (FFN) masked weights
    const float* imsL = ims + (size_t)l * 3 * FF * DD;
    k_sel_init<<<3, 256, 0, stream>>>(selState, hist, (u32)((size_t)FF * DD / 2));
    for (int shift = 24; shift >= 0; shift -= 8) {
      k_sel_hist<<<dim3(1024, 3), 256, 0, stream>>>(imsL, (u32)((size_t)FF * DD),
                                                    shift, selState, hist);
      k_sel_scan<<<3, 1, 0, stream>>>(shift, selState, hist, thr, 9);
    }
    k_maskw<<<4096, 256, 0, stream>>>(iW + (size_t)l * FF * DD, imsL, thr, 9,
                                      mwi, (size_t)FF * DD);
    k_cvt_bf<<<1024, 256, 0, stream>>>(aoW + (size_t)l * DD * DD, wao,
                                       (size_t)DD * DD);
    k_cvt_bf<<<2048, 256, 0, stream>>>(oW + (size_t)l * DD * FF, wo,
                                       (size_t)DD * FF);

    // attention
    k_vt<<<2048, 256, 0, stream>>>(v_bf, vtb);
    k_attn<<<dim3(SS / 16, BB * NH), 32, 0, stream>>>(
        q_bf, kk_bf, vtb, amask, hmask + (size_t)l * NH, ctx_bf);
    k_gemm<1><<<gemmD, 256, 0, stream>>>(ctx_bf, wao, aob + (size_t)l * DD,
                                         nullptr, tmp, nullptr, DD, DD);
    k_ln<<<MM, 256, 0, stream>>>(tmp, hin, ln1g + (size_t)l * DD,
                                 ln1b + (size_t)l * DD, attn, ao_bf);

    // FFN
    k_gate3<<<MM, 256, 0, stream>>>(attn, igw + (size_t)l * 3 * DD,
                                    igb + (size_t)l * 3, expi);
    k_gemm<3><<<gemmF, 256, 0, stream>>>(ao_bf, mwi, ib_ + (size_t)l * FF, expi,
                                         inter, nullptr, FF, DD);
    k_gelu_bf<<<4096, 256, 0, stream>>>(inter, int_bf, (size_t)MM * FF);
    k_gemm<1><<<gemmD, 256, 0, stream>>>(int_bf, wo, ob_ + (size_t)l * DD,
                                         nullptr, tmp, nullptr, DD, FF);
    float* outp = (l == LAYERS - 1) ? (float*)d_out : hcur;
    k_ln<<<MM, 256, 0, stream>>>(tmp, attn, ln2g + (size_t)l * DD,
                                 ln2b + (size_t)l * DD, outp, nullptr);
  }
}
